// covnext_tiny_attention_output_83141976916043
// MI455X (gfx1250) — matrix operand order repaired and checked by reading, not compiled, not run on hardware
//
#include <hip/hip_runtime.h>
#include <hip/hip_bf16.h>

// CDNA5 / gfx1250: wave32, WMMA 16x16x32 bf16 (f32 accum).
typedef __attribute__((ext_vector_type(16))) __bf16 v16bf;
typedef __attribute__((ext_vector_type(8)))  float  v8f;

#define HDIM  768
#define BATCH 32
#define TSTEPS 50
#define VOCAB 32000
#define NPOS  49
#define NT    4      // N tiles per wave (4 waves/block => 256 cols per block)

__device__ __forceinline__ float sigf(float x) { return 1.f / (1.f + __expf(-x)); }

// ---- WMMA fragment loads from precomputed per-lane row pointers ----------
// A (16x32 MxK): lane row = m0 + (lane&15); kb = (lane>>4)*16.
// f[0..15] = row[k0+kb .. +15]  (2x b128)
__device__ __forceinline__ v16bf frag_a(const __bf16* __restrict__ arow, int kb, int k0) {
    v16bf f;
#pragma unroll
    for (int j = 0; j < 16; ++j) f[j] = arow[k0 + kb + j];
    return f;
}
// B (32x16 KxN from W[N,K] row-major): lane col = n0 + (lane&15),
// K window base = (lane>>4)*16; 16 contiguous bf16 (2x b128).
__device__ __forceinline__ v16bf frag_b(const __bf16* __restrict__ brow, int k0) {
    v16bf f;
#pragma unroll
    for (int j = 0; j < 16; ++j) f[j] = brow[k0 + j];
    return f;
}

#define WMMA_BF16(ACC, Afr, Bfr) \
    ACC = __builtin_amdgcn_wmma_f32_16x16x32_bf16(false, Afr, false, Bfr, (short)0, ACC, false, false)

// ---- Generic bf16 WMMA GEMM: C = act(A @ W^T + bias) --------------------
// A: [M,K] bf16 (lda), W: [N,K] bf16 (ldw), C: f32 (ldc).
// Requires N % 256 == 0, K % 128 == 0, M % 16 == 0 (true for all call sites).
// mode 0: plain   mode 1: tanh   mode 2: store out[b][n][t] with m = b*Td + t
// k-loop: chunked software pipeline. Outer loop strides 128 in K and issues
// one unconditional prefetch burst; inner (fully unrolled) preloads step i+1
// before the 4 WMMAs of step i, so waits are partial and loads stay in flight.
__global__ __launch_bounds__(128) void gemm_bf16_wmma(
    const __bf16* __restrict__ A, int lda,
    const __bf16* __restrict__ W, int ldw,
    const float* __restrict__ bias,
    float* __restrict__ C, int ldc,
    int M, int N, int K, int mode, int Td)
{
    const int lane = threadIdx.x & 31;
    const int wave = threadIdx.x >> 5;
    const int n0 = (blockIdx.x * 4 + wave) * (NT * 16);  // 64 cols per wave
    const int m0 = blockIdx.y * 16;
    if (m0 >= M || n0 >= N) return;                      // wave-uniform

    // per-lane hoisted row pointers
    const int akb = (lane >> 4) << 4;
    const __bf16* arow = A + (size_t)(m0 + (lane & 15)) * lda;
    const __bf16* b0r = W + (size_t)(n0 +  0 + (lane & 15)) * ldw + ((lane >> 4) << 4);
    const __bf16* b1r = W + (size_t)(n0 + 16 + (lane & 15)) * ldw + ((lane >> 4) << 4);
    const __bf16* b2r = W + (size_t)(n0 + 32 + (lane & 15)) * ldw + ((lane >> 4) << 4);
    const __bf16* b3r = W + (size_t)(n0 + 48 + (lane & 15)) * ldw + ((lane >> 4) << 4);

    v8f acc0 = {}, acc1 = {}, acc2 = {}, acc3 = {};

    // pipeline prologue: fragments for k = 0
    v16bf a  = frag_a(arow, akb, 0);
    v16bf b0 = frag_b(b0r, 0);
    v16bf b1 = frag_b(b1r, 0);
    v16bf b2 = frag_b(b2r, 0);
    v16bf b3 = frag_b(b3r, 0);

    // all 128-chunks except the last; current frags enter at k0, leave at k0+128
    int k0 = 0;
    for (; k0 + 256 <= K; k0 += 128) {
        __builtin_prefetch(b0r + k0 + 256, 0, 3);
        __builtin_prefetch(b1r + k0 + 256, 0, 3);
        __builtin_prefetch(b2r + k0 + 256, 0, 3);
        __builtin_prefetch(b3r + k0 + 256, 0, 3);
        __builtin_prefetch(arow + k0 + 256, 0, 3);
#pragma unroll
        for (int kk = 32; kk <= 128; kk += 32) {
            v16bf an  = frag_a(arow, akb, k0 + kk);
            v16bf b0n = frag_b(b0r, k0 + kk);
            v16bf b1n = frag_b(b1r, k0 + kk);
            v16bf b2n = frag_b(b2r, k0 + kk);
            v16bf b3n = frag_b(b3r, k0 + kk);
            WMMA_BF16(acc0, a, b0);
            WMMA_BF16(acc1, a, b1);
            WMMA_BF16(acc2, a, b2);
            WMMA_BF16(acc3, a, b3);
            a = an; b0 = b0n; b1 = b1n; b2 = b2n; b3 = b3n;
        }
    }
    // final 128-chunk (k0 == K-128): 3 pipelined sub-steps + drain
#pragma unroll
    for (int kk = 32; kk <= 96; kk += 32) {
        v16bf an  = frag_a(arow, akb, k0 + kk);
        v16bf b0n = frag_b(b0r, k0 + kk);
        v16bf b1n = frag_b(b1r, k0 + kk);
        v16bf b2n = frag_b(b2r, k0 + kk);
        v16bf b3n = frag_b(b3r, k0 + kk);
        WMMA_BF16(acc0, a, b0);
        WMMA_BF16(acc1, a, b1);
        WMMA_BF16(acc2, a, b2);
        WMMA_BF16(acc3, a, b3);
        a = an; b0 = b0n; b1 = b1n; b2 = b2n; b3 = b3n;
    }
    WMMA_BF16(acc0, a, b0);
    WMMA_BF16(acc1, a, b1);
    WMMA_BF16(acc2, a, b2);
    WMMA_BF16(acc3, a, b3);

    // C/D layout: lane&15 = column, VGPR r = row r (lanes 0-15) / r+8 (16-31)
    const v8f accs[NT] = { acc0, acc1, acc2, acc3 };
    const int mh = (lane >> 4) << 3;
#pragma unroll
    for (int i = 0; i < NT; ++i) {
        const int col = n0 + 16 * i + (lane & 15);
        const float bv = bias ? bias[col] : 0.f;
#pragma unroll
        for (int r = 0; r < 8; ++r) {
            int row = m0 + mh + r;
            if (row >= M) continue;
            float v = accs[i][r] + bv;
            if (mode == 1) v = tanhf(v);
            if (mode == 2) {
                int b_ = row / Td, t_ = row % Td;
                C[(size_t)b_ * ldc * Td + (size_t)col * Td + t_] = v;
            } else {
                C[(size_t)row * ldc + col] = v;
            }
        }
    }
}

// ---- weight prep ---------------------------------------------------------
__global__ void concat_weights_kernel(const float* __restrict__ wih,
                                      const float* __restrict__ whh,
                                      const float* __restrict__ bih,
                                      const float* __restrict__ bhh,
                                      __bf16* __restrict__ wcat,
                                      float* __restrict__ bcat)
{
    int n = blockIdx.x;  // 0..4H-1
    for (int k = threadIdx.x; k < 2 * HDIM; k += blockDim.x) {
        float v = (k < HDIM) ? wih[(size_t)n * HDIM + k]
                             : whh[(size_t)n * HDIM + (k - HDIM)];
        wcat[(size_t)n * (2 * HDIM) + k] = (__bf16)v;
    }
    if (threadIdx.x == 0) bcat[n] = bih[n] + bhh[n];
}

__global__ void f32_to_bf16_kernel(const float* __restrict__ src,
                                   __bf16* __restrict__ dst, size_t n)
{
    size_t i = (size_t)blockIdx.x * blockDim.x + threadIdx.x;
    if (i < n) dst[i] = (__bf16)src[i];
}

// fmaps [B,H,49] -> fm_bf [B,49,H]
__global__ void fm_transpose_kernel(const float* __restrict__ fmaps,
                                    __bf16* __restrict__ fm_bf)
{
    int b = blockIdx.x, p = blockIdx.y;
    for (int h = threadIdx.x; h < HDIM; h += blockDim.x)
        fm_bf[((size_t)b * NPOS + p) * HDIM + h] =
            (__bf16)fmaps[((size_t)b * HDIM + h) * NPOS + p];
}

// avgpool, state init, x0 = embed[sos], pred0 = [avg | x0]
__global__ void init_kernel(const float* __restrict__ fmaps,
                            const float* __restrict__ embed,
                            const int* __restrict__ sos,
                            float* __restrict__ h1, float* __restrict__ c0,
                            float* __restrict__ c1,
                            __bf16* __restrict__ xh0_bf, __bf16* __restrict__ xh1_bf,
                            __bf16* __restrict__ predTB, __bf16* __restrict__ predBT)
{
    int b = blockIdx.x;
    int s = *sos;
    for (int h = threadIdx.x; h < HDIM; h += blockDim.x) {
        const float* f = fmaps + ((size_t)b * HDIM + h) * NPOS;
        float a = 0.f;
        for (int p = 0; p < NPOS; ++p) a += f[p];
        a *= (1.f / (float)NPOS);
        h1[b * HDIM + h] = a;
        c0[b * HDIM + h] = 0.f;
        c1[b * HDIM + h] = 0.f;
        float x0 = embed[(size_t)s * HDIM + h];
        __bf16 ab = (__bf16)a, xb = (__bf16)x0;
        xh0_bf[(size_t)b * 2 * HDIM + h] = xb;          // x
        xh0_bf[(size_t)b * 2 * HDIM + HDIM + h] = ab;   // h0 = avg
        xh1_bf[(size_t)b * 2 * HDIM + HDIM + h] = ab;   // h1 = avg
        size_t rTB = (size_t)b * 2 * HDIM;              // t = 0
        predTB[rTB + h] = ab; predTB[rTB + HDIM + h] = xb;
        size_t rBT = (size_t)b * TSTEPS * 2 * HDIM;
        predBT[rBT + h] = ab; predBT[rBT + HDIM + h] = xb;
    }
}

// ---- LSTM cell (gate order i,f,g,o) -------------------------------------
__global__ void lstm_cell_kernel(const float* __restrict__ gates,  // [B,4H]
                                 float* __restrict__ c,            // [B,H] in/out
                                 float* __restrict__ h_f32,        // [B,H] out
                                 __bf16* __restrict__ h_bf_a, int strideA,
                                 __bf16* __restrict__ h_bf_b, int strideB)
{
    int b = blockIdx.x;
    const float* g = gates + (size_t)b * 4 * HDIM;
    for (int h = threadIdx.x; h < HDIM; h += blockDim.x) {
        float ig = sigf(g[h]);
        float fg = sigf(g[HDIM + h]);
        float gg = tanhf(g[2 * HDIM + h]);
        float og = sigf(g[3 * HDIM + h]);
        float cn = fg * c[b * HDIM + h] + ig * gg;
        c[b * HDIM + h] = cn;
        float hn = og * tanhf(cn);
        h_f32[b * HDIM + h] = hn;
        __bf16 hb = (__bf16)hn;
        h_bf_a[(size_t)b * strideA + h] = hb;
        h_bf_b[(size_t)b * strideB + h] = hb;
    }
}

// ---- attention + pred write ---------------------------------------------
__global__ void attention_kernel(const float* __restrict__ keys,    // [B,49,H]
                                 const float* __restrict__ values,  // [B,49,H]
                                 const float* __restrict__ query,   // [B,H]
                                 const float* __restrict__ h1,      // [B,H]
                                 __bf16* __restrict__ predTB,
                                 __bf16* __restrict__ predBT, int t)
{
    __shared__ float red[256];
    __shared__ float sc[NPOS];
    __shared__ float ssum;
    int b = blockIdx.x;
    const float* q = query + (size_t)b * HDIM;
    for (int p = 0; p < NPOS; ++p) {
        const float* kr = keys + ((size_t)b * NPOS + p) * HDIM;
        float s = 0.f;
        for (int h = threadIdx.x; h < HDIM; h += 256) s += kr[h] * q[h];
        red[threadIdx.x] = s;
        __syncthreads();
        for (int off = 128; off > 0; off >>= 1) {
            if (threadIdx.x < off) red[threadIdx.x] += red[threadIdx.x + off];
            __syncthreads();
        }
        if (threadIdx.x == 0) sc[p] = red[0];
        __syncthreads();
    }
    if (threadIdx.x == 0) {
        float mx = sc[0];
        for (int p = 1; p < NPOS; ++p) mx = fmaxf(mx, sc[p]);
        float su = 0.f;
        for (int p = 0; p < NPOS; ++p) { sc[p] = __expf(sc[p] - mx); su += sc[p]; }
        ssum = su;
    }
    __syncthreads();
    float inv = 1.f / ssum;
    for (int h = threadIdx.x; h < HDIM; h += 256) {
        float ctx = 0.f;
        for (int p = 0; p < NPOS; ++p)
            ctx += sc[p] * values[((size_t)b * NPOS + p) * HDIM + h];
        ctx *= inv;
        __bf16 cb = (__bf16)ctx;
        __bf16 hb = (__bf16)h1[(size_t)b * HDIM + h];
        size_t rTB = ((size_t)t * BATCH + b) * 2 * HDIM;
        predTB[rTB + h] = cb; predTB[rTB + HDIM + h] = hb;
        size_t rBT = ((size_t)b * TSTEPS + t) * 2 * HDIM;
        predBT[rBT + h] = cb; predBT[rBT + HDIM + h] = hb;
    }
}

// ---- argmax over vocab + embedding gather into next x -------------------
__global__ void argmax_gather_kernel(const float* __restrict__ logits, // [B,V]
                                     const float* __restrict__ embed,  // [V,H]
                                     __bf16* __restrict__ x_bf, int xstride)
{
    __shared__ float bv[256];
    __shared__ int   bi[256];
    int b = blockIdx.x;
    const float* lg = logits + (size_t)b * VOCAB;
    float best = -3.0e38f; int bidx = 0;
    for (int v = threadIdx.x; v < VOCAB; v += 256) {
        float val = lg[v];
        if (val > best) { best = val; bidx = v; }
    }
    bv[threadIdx.x] = best; bi[threadIdx.x] = bidx;
    __syncthreads();
    for (int off = 128; off > 0; off >>= 1) {
        if (threadIdx.x < off) {
            float ov = bv[threadIdx.x + off]; int oi = bi[threadIdx.x + off];
            if (ov > bv[threadIdx.x] || (ov == bv[threadIdx.x] && oi < bi[threadIdx.x])) {
                bv[threadIdx.x] = ov; bi[threadIdx.x] = oi;
            }
        }
        __syncthreads();
    }
    int idx = bi[0];
    const float* er = embed + (size_t)idx * HDIM;
    for (int h = threadIdx.x; h < HDIM; h += 256)
        x_bf[(size_t)b * xstride + h] = (__bf16)er[h];
}

// =========================================================================
extern "C" void kernel_launch(void* const* d_in, const int* in_sizes, int n_in,
                              void* d_out, int out_size, void* d_ws, size_t ws_size,
                              hipStream_t stream)
{
    const float* fmaps  = (const float*)d_in[0];
    const float* embed  = (const float*)d_in[1];
    const float* Wq = (const float*)d_in[2];  const float* bq = (const float*)d_in[3];
    const float* Wk = (const float*)d_in[4];  const float* bk = (const float*)d_in[5];
    const float* Wv = (const float*)d_in[6];  const float* bv = (const float*)d_in[7];
    const float* w_ih0 = (const float*)d_in[8];  const float* w_hh0 = (const float*)d_in[9];
    const float* b_ih0 = (const float*)d_in[10]; const float* b_hh0 = (const float*)d_in[11];
    const float* w_ih1 = (const float*)d_in[12]; const float* w_hh1 = (const float*)d_in[13];
    const float* b_ih1 = (const float*)d_in[14]; const float* b_hh1 = (const float*)d_in[15];
    const float* proj_w = (const float*)d_in[16];
    const float* proj_b = (const float*)d_in[17];
    const int*   sos    = (const int*)d_in[19];
    float* out = (float*)d_out;

    // ---- workspace carving ----
    char* ws = (char*)d_ws;
    size_t off = 0;
    auto carve = [&](size_t bytes) {
        void* p = ws + off;
        off = (off + bytes + 255) & ~(size_t)255;
        return p;
    };
    __bf16* wcat0    = (__bf16*)carve((size_t)4 * HDIM * 2 * HDIM * 2);
    __bf16* wcat1    = (__bf16*)carve((size_t)4 * HDIM * 2 * HDIM * 2);
    __bf16* wq_bf    = (__bf16*)carve((size_t)HDIM * HDIM * 2);
    __bf16* wk_bf    = (__bf16*)carve((size_t)HDIM * HDIM * 2);
    __bf16* wv_bf    = (__bf16*)carve((size_t)HDIM * HDIM * 2);
    __bf16* projw_bf = (__bf16*)carve((size_t)VOCAB * 2 * HDIM * 2);   // 98 MB: L2-resident
    __bf16* fm_bf    = (__bf16*)carve((size_t)BATCH * NPOS * HDIM * 2);
    float*  keys     = (float*)carve((size_t)BATCH * NPOS * HDIM * 4);
    float*  values   = (float*)carve((size_t)BATCH * NPOS * HDIM * 4);
    __bf16* predTB   = (__bf16*)carve((size_t)TSTEPS * BATCH * 2 * HDIM * 2);
    __bf16* predBT   = (__bf16*)carve((size_t)BATCH * TSTEPS * 2 * HDIM * 2);
    float*  logits   = (float*)carve((size_t)BATCH * VOCAB * 4);
    float*  gates    = (float*)carve((size_t)BATCH * 4 * HDIM * 4);
    float*  h0s      = (float*)carve((size_t)BATCH * HDIM * 4);
    float*  h1s      = (float*)carve((size_t)BATCH * HDIM * 4);
    float*  c0s      = (float*)carve((size_t)BATCH * HDIM * 4);
    float*  c1s      = (float*)carve((size_t)BATCH * HDIM * 4);
    float*  query    = (float*)carve((size_t)BATCH * HDIM * 4);
    __bf16* h1n_bf   = (__bf16*)carve((size_t)BATCH * HDIM * 2);
    __bf16* xh0_bf   = (__bf16*)carve((size_t)BATCH * 2 * HDIM * 2);
    __bf16* xh1_bf   = (__bf16*)carve((size_t)BATCH * 2 * HDIM * 2);
    float*  bcat0    = (float*)carve((size_t)4 * HDIM * 4);
    float*  bcat1    = (float*)carve((size_t)4 * HDIM * 4);
    (void)ws_size; (void)in_sizes; (void)n_in; (void)out_size;

    // ---- preprocess ----
    concat_weights_kernel<<<4 * HDIM, 256, 0, stream>>>(w_ih0, w_hh0, b_ih0, b_hh0, wcat0, bcat0);
    concat_weights_kernel<<<4 * HDIM, 256, 0, stream>>>(w_ih1, w_hh1, b_ih1, b_hh1, wcat1, bcat1);
    {
        size_t n = (size_t)HDIM * HDIM;
        int g = (int)((n + 255) / 256);
        f32_to_bf16_kernel<<<g, 256, 0, stream>>>(Wq, wq_bf, n);
        f32_to_bf16_kernel<<<g, 256, 0, stream>>>(Wk, wk_bf, n);
        f32_to_bf16_kernel<<<g, 256, 0, stream>>>(Wv, wv_bf, n);
        size_t np = (size_t)VOCAB * 2 * HDIM;
        f32_to_bf16_kernel<<<(int)((np + 255) / 256), 256, 0, stream>>>(proj_w, projw_bf, np);
    }
    fm_transpose_kernel<<<dim3(BATCH, NPOS), 256, 0, stream>>>(fmaps, fm_bf);

    // keys/values: tanh(fm @ W^T + b)   M=1568, N=768, K=768
    gemm_bf16_wmma<<<dim3(HDIM / 256, (BATCH * NPOS) / 16), 128, 0, stream>>>(
        fm_bf, HDIM, wk_bf, HDIM, bk, keys, HDIM, BATCH * NPOS, HDIM, HDIM, 1, 0);
    gemm_bf16_wmma<<<dim3(HDIM / 256, (BATCH * NPOS) / 16), 128, 0, stream>>>(
        fm_bf, HDIM, wv_bf, HDIM, bv, values, HDIM, BATCH * NPOS, HDIM, HDIM, 1, 0);

    init_kernel<<<BATCH, 256, 0, stream>>>(fmaps, embed, sos, h1s, c0s, c1s,
                                           xh0_bf, xh1_bf, predTB, predBT);

    // ---- sequential decode: 49 steps ----
    for (int t = 0; t < TSTEPS - 1; ++t) {
        // layer0 gates: [32,1536] @ [3072,1536]^T
        gemm_bf16_wmma<<<dim3((4 * HDIM) / 256, BATCH / 16), 128, 0, stream>>>(
            xh0_bf, 2 * HDIM, wcat0, 2 * HDIM, bcat0, gates, 4 * HDIM,
            BATCH, 4 * HDIM, 2 * HDIM, 0, 0);
        // cell0 -> h0n into xh1[:H] (this step) and xh0[H:] (next step)
        lstm_cell_kernel<<<BATCH, 256, 0, stream>>>(gates, c0s, h0s,
            xh1_bf, 2 * HDIM, xh0_bf + HDIM, 2 * HDIM);
        // layer1 gates
        gemm_bf16_wmma<<<dim3((4 * HDIM) / 256, BATCH / 16), 128, 0, stream>>>(
            xh1_bf, 2 * HDIM, wcat1, 2 * HDIM, bcat1, gates, 4 * HDIM,
            BATCH, 4 * HDIM, 2 * HDIM, 0, 0);
        // cell1 -> h1n into h1n_bf (query) and xh1[H:] (next step)
        lstm_cell_kernel<<<BATCH, 256, 0, stream>>>(gates, c1s, h1s,
            h1n_bf, HDIM, xh1_bf + HDIM, 2 * HDIM);
        // query = tanh(h1n @ Wq^T + bq)
        gemm_bf16_wmma<<<dim3(HDIM / 256, BATCH / 16), 128, 0, stream>>>(
            h1n_bf, HDIM, wq_bf, HDIM, bq, query, HDIM, BATCH, HDIM, HDIM, 1, 0);
        // attention, writes pred[t+1] in both layouts
        attention_kernel<<<BATCH, 256, 0, stream>>>(keys, values, query, h1s,
                                                    predTB, predBT, t + 1);
        // logits = pred[t+1] @ proj_w^T + proj_b   (proj_w bf16 = 98MB, L2-resident)
        gemm_bf16_wmma<<<dim3(VOCAB / 256, BATCH / 16), 128, 0, stream>>>(
            predTB + (size_t)(t + 1) * BATCH * 2 * HDIM, 2 * HDIM,
            projw_bf, 2 * HDIM, proj_b, logits, VOCAB,
            BATCH, VOCAB, 2 * HDIM, 0, 0);
        // argmax -> x_{t+1} = embed[idx] into xh0[:H]
        argmax_gather_kernel<<<BATCH, 256, 0, stream>>>(logits, embed, xh0_bf, 2 * HDIM);
    }

    // ---- final projection: [B*T,2H] @ [V,2H]^T -> out[B,V,T] ----
    gemm_bf16_wmma<<<dim3(VOCAB / 256, (BATCH * TSTEPS) / 16), 128, 0, stream>>>(
        predBT, 2 * HDIM, projw_bf, 2 * HDIM, proj_b, out, VOCAB,
        BATCH * TSTEPS, VOCAB, 2 * HDIM, 2, TSTEPS);
}
